// UniGCNII_4088808866004
// MI455X (gfx1250) — matrix operand order repaired and checked by reading, not compiled, not run on hardware
//
#include <hip/hip_runtime.h>

typedef __attribute__((ext_vector_type(16))) _Float16 v16h;
typedef __attribute__((ext_vector_type(8)))  _Float16 v8h;
typedef __attribute__((ext_vector_type(8)))  float    v8f;

#define NV      100000
#define NEDGE   20000
#define NNZP    1600000
#define NFEAT   128
#define NHID    64
#define NCLASS  40
#define NLAYER  4
#define NPAD    48          // classes padded to 3 tiles of 16
#define MTILES  (NV/16)     // 6250 exact

// ------------------------------ prep kernels ------------------------------

__global__ void k_cvt_f16(const float* __restrict__ src, _Float16* __restrict__ dst, int n) {
  int i = blockIdx.x * blockDim.x + threadIdx.x;
  if (i < n) dst[i] = (_Float16)src[i];
}

// weights -> f16, transposed to [N][K] so WMMA B loads are contiguous per lane
__global__ void k_prep_w(const float* __restrict__ W0, const float* __restrict__ Ws,
                         const float* __restrict__ Wout,
                         _Float16* __restrict__ w0t, _Float16* __restrict__ wst,
                         _Float16* __restrict__ woutt) {
  int i = blockIdx.x * blockDim.x + threadIdx.x;
  const int n0 = NHID * NFEAT;          // 8192
  const int n1 = NLAYER * NHID * NHID;  // 16384
  const int n2 = NPAD * NHID;           // 3072
  if (i < n0) {
    int n = i / NFEAT, k = i % NFEAT;
    w0t[i] = (_Float16)W0[k * NHID + n];
  } else if (i < n0 + n1) {
    int j = i - n0;
    int l = j / (NHID * NHID), r = j % (NHID * NHID);
    int n = r / NHID, k = r % NHID;
    wst[j] = (_Float16)Ws[l * NHID * NHID + k * NHID + n];
  } else if (i < n0 + n1 + n2) {
    int j = i - n0 - n1;
    int n = j / NHID, k = j % NHID;
    woutt[j] = (n < NCLASS) ? (_Float16)Wout[k * NCLASS + n] : (_Float16)0.0f;
  }
}

__global__ void k_count(const int* __restrict__ edges, float* __restrict__ cnt) {
  int i = blockIdx.x * blockDim.x + threadIdx.x;
  if (i < NNZP) atomicAdd(&cnt[edges[i]], 1.0f);
}

__global__ void k_escale(const float* __restrict__ degE, float* __restrict__ esc) {
  int e = blockIdx.x * blockDim.x + threadIdx.x;
  if (e < NEDGE) esc[e] = degE[e] / fmaxf(esc[e], 1.0f);
}

// ------------------------------ WMMA core ------------------------------
// A: f16 row-major [M][K]. Bt: f16 [N][K] (i.e. B transposed).
// A as loaded here (not AMD's table order): lane (row = lane&15, half = lane>>4) holds
// K = k0 + 16*half .. +15 (contiguous, matching B).
// B layout: lane holds column N = nt*16 + (lane&15), K = k0 + 16*half .. +15 (contiguous in Bt).
template <int K, int NT>
__device__ __forceinline__ void wmma_rows(const _Float16* __restrict__ A,
                                          const _Float16* __restrict__ Bt,
                                          int m0, int lane, v8f c[NT]) {
  const int row  = lane & 15;
  const int half = lane >> 4;
#pragma unroll
  for (int kt = 0; kt < K / 32; ++kt) {
    const int k0 = kt * 32;
    union { v16h v; v8h h[2]; } a;
    const _Float16* ap = A + (size_t)(m0 + row) * K + k0 + 16 * half;
    a.h[0] = *(const v8h*)(ap);
    a.h[1] = *(const v8h*)(ap + 8);
#pragma unroll
    for (int nt = 0; nt < NT; ++nt) {
      const _Float16* bp = Bt + (size_t)(nt * 16 + row) * K + k0 + 16 * half;
      v16h b = *(const v16h*)(bp);
      c[nt] = __builtin_amdgcn_wmma_f32_16x16x32_f16(false, a.v, false, b,
                                                     (short)0, c[nt], false, false);
    }
  }
}

// X0 = relu(x @ W0 + b0); write fp32 twice (X0 residual + current X)
__global__ void k_gemm_in(const _Float16* __restrict__ xh, const _Float16* __restrict__ w0t,
                          const float* __restrict__ b0, float* __restrict__ x0,
                          float* __restrict__ xcur) {
  const int lane = threadIdx.x & 31;
  const int tile = blockIdx.x * 4 + (threadIdx.x >> 5);
  if (tile >= MTILES) return;  // wave-uniform
  v8f c[4] = {};
  wmma_rows<NFEAT, 4>(xh, w0t, tile * 16, lane, c);
  const int row = lane & 15, half = lane >> 4;
#pragma unroll
  for (int nt = 0; nt < 4; ++nt)
#pragma unroll
    for (int r = 0; r < 8; ++r) {
      int mm = tile * 16 + r + 8 * half;
      int nn = nt * 16 + row;
      float v = fmaxf(c[nt][r] + b0[nn], 0.0f);
      size_t o = (size_t)mm * NHID + nn;
      x0[o] = v;
      xcur[o] = v;
    }
}

// X = relu((1-beta)*Xi + beta*(Xi @ Ws[l])); also emit f16 copy for next GEMM
__global__ void k_gemm_layer(const _Float16* __restrict__ xih, const _Float16* __restrict__ wt,
                             const float* __restrict__ xi, float beta,
                             float* __restrict__ xcur, _Float16* __restrict__ xhout) {
  const int lane = threadIdx.x & 31;
  const int tile = blockIdx.x * 4 + (threadIdx.x >> 5);
  if (tile >= MTILES) return;
  v8f c[4] = {};
  wmma_rows<NHID, 4>(xih, wt, tile * 16, lane, c);
  const int row = lane & 15, half = lane >> 4;
  const float ob = 1.0f - beta;
#pragma unroll
  for (int nt = 0; nt < 4; ++nt)
#pragma unroll
    for (int r = 0; r < 8; ++r) {
      int mm = tile * 16 + r + 8 * half;
      int nn = nt * 16 + row;
      size_t o = (size_t)mm * NHID + nn;
      float v = fmaxf(ob * xi[o] + beta * c[nt][r], 0.0f);
      xcur[o] = v;
      xhout[o] = (_Float16)v;
    }
}

// logits = X @ Wout + bout  (N padded 40->48, mask stores)
__global__ void k_gemm_out(const _Float16* __restrict__ xh, const _Float16* __restrict__ woutt,
                           const float* __restrict__ bout, float* __restrict__ out) {
  const int lane = threadIdx.x & 31;
  const int tile = blockIdx.x * 4 + (threadIdx.x >> 5);
  if (tile >= MTILES) return;
  v8f c[3] = {};
  wmma_rows<NHID, 3>(xh, woutt, tile * 16, lane, c);
  const int row = lane & 15, half = lane >> 4;
#pragma unroll
  for (int nt = 0; nt < 3; ++nt)
#pragma unroll
    for (int r = 0; r < 8; ++r) {
      int mm = tile * 16 + r + 8 * half;
      int nn = nt * 16 + row;
      if (nn < NCLASS) out[(size_t)mm * NCLASS + nn] = c[nt][r] + bout[nn];
    }
}

// ------------------------------ aggregation ------------------------------

// dst[didx[p]] += src[sidx[p]]   (rows of 64 floats; 1 thread per element)
__global__ void k_scatter(const float* __restrict__ src, const int* __restrict__ sidx,
                          const int* __restrict__ didx, float* __restrict__ dst) {
  int i = blockIdx.x * blockDim.x + threadIdx.x;  // < NNZ*64
  if (i < NNZP * NHID) {
    int p = i >> 6, f = i & 63;
    atomicAdd(&dst[(size_t)didx[p] * NHID + f], src[(size_t)sidx[p] * NHID + f]);
  }
}

__global__ void k_edge_scale(float* __restrict__ xe, const float* __restrict__ esc) {
  int i = blockIdx.x * blockDim.x + threadIdx.x;
  if (i < NEDGE * NHID) xe[i] *= esc[i >> 6];
}

// per-vertex: scale by degV, L2-normalize, mix with X0 -> Xi (in place) + f16 copy
__global__ void k_vertex_update(float* __restrict__ xv, const float* __restrict__ degV,
                                const float* __restrict__ x0, _Float16* __restrict__ xih) {
  const int lane = threadIdx.x & 31;
  const int rowv = blockIdx.x * 8 + (threadIdx.x >> 5);
  if (rowv >= NV) return;
  size_t b = (size_t)rowv * NHID;
  float dv = degV[rowv];
  float a0 = xv[b + lane] * dv;
  float a1 = xv[b + 32 + lane] * dv;
  float ss = a0 * a0 + a1 * a1;
#pragma unroll
  for (int off = 16; off; off >>= 1) ss += __shfl_xor(ss, off, 32);
  float sc = (ss > 0.0f) ? (1.0f / sqrtf(ss)) : 0.0f;
  float v0 = 0.9f * (a0 * sc) + 0.1f * x0[b + lane];
  float v1 = 0.9f * (a1 * sc) + 0.1f * x0[b + 32 + lane];
  xv[b + lane] = v0;
  xv[b + 32 + lane] = v1;
  xih[b + lane] = (_Float16)v0;
  xih[b + 32 + lane] = (_Float16)v1;
}

__global__ void k_logsoftmax(float* __restrict__ out) {
  const int lane = threadIdx.x & 31;
  const int rowv = blockIdx.x * 8 + (threadIdx.x >> 5);
  if (rowv >= NV) return;
  size_t b = (size_t)rowv * NCLASS;
  float v0 = (lane < NCLASS) ? out[b + lane] : -__builtin_inff();
  float v1 = (lane + 32 < NCLASS) ? out[b + 32 + lane] : -__builtin_inff();
  float mx = fmaxf(v0, v1);
#pragma unroll
  for (int off = 16; off; off >>= 1) mx = fmaxf(mx, __shfl_xor(mx, off, 32));
  float s = ((lane < NCLASS) ? expf(v0 - mx) : 0.0f) +
            ((lane + 32 < NCLASS) ? expf(v1 - mx) : 0.0f);
#pragma unroll
  for (int off = 16; off; off >>= 1) s += __shfl_xor(s, off, 32);
  float lse = mx + logf(s);
  if (lane < NCLASS) out[b + lane] = v0 - lse;
  if (lane + 32 < NCLASS) out[b + 32 + lane] = v1 - lse;
}

// ------------------------------ host ------------------------------

extern "C" void kernel_launch(void* const* d_in, const int* in_sizes, int n_in,
                              void* d_out, int out_size, void* d_ws, size_t ws_size,
                              hipStream_t stream) {
  const float* x    = (const float*)d_in[0];
  const float* degE = (const float*)d_in[1];
  const float* degV = (const float*)d_in[2];
  const float* W0   = (const float*)d_in[3];
  const float* b0   = (const float*)d_in[4];
  const float* Ws   = (const float*)d_in[5];
  const float* Wout = (const float*)d_in[6];
  const float* bout = (const float*)d_in[7];
  const int* vertex = (const int*)d_in[8];
  const int* edges  = (const int*)d_in[9];
  float* out = (float*)d_out;
  (void)in_sizes; (void)n_in; (void)out_size; (void)ws_size;

  char* ws = (char*)d_ws;
  size_t off = 0;
  auto take = [&](size_t bytes) -> char* {
    char* p = ws + off;
    off = (off + bytes + 255) & ~(size_t)255;
    return p;
  };
  _Float16* xh    = (_Float16*)take((size_t)NV * NFEAT * 2);
  _Float16* w0t   = (_Float16*)take((size_t)NHID * NFEAT * 2);
  _Float16* wst   = (_Float16*)take((size_t)NLAYER * NHID * NHID * 2);
  _Float16* woutt = (_Float16*)take((size_t)NPAD * NHID * 2);
  float*    esc   = (float*)take((size_t)NEDGE * 4);
  float*    x0    = (float*)take((size_t)NV * NHID * 4);
  float*    xcur  = (float*)take((size_t)NV * NHID * 4);
  float*    xv    = (float*)take((size_t)NV * NHID * 4);   // Xv, then Xi (in place)
  _Float16* xih   = (_Float16*)take((size_t)NV * NHID * 2);
  _Float16* xfh   = (_Float16*)take((size_t)NV * NHID * 2);
  float*    xe    = (float*)take((size_t)NEDGE * NHID * 4);

  // edge scale = degE / max(count,1)
  hipMemsetAsync(esc, 0, (size_t)NEDGE * sizeof(float), stream);
  k_count<<<(NNZP + 255) / 256, 256, 0, stream>>>(edges, esc);
  k_escale<<<(NEDGE + 255) / 256, 256, 0, stream>>>(degE, esc);

  // f16 conversions
  k_cvt_f16<<<(NV * NFEAT + 255) / 256, 256, 0, stream>>>(x, xh, NV * NFEAT);
  k_prep_w<<<((NHID * NFEAT + NLAYER * NHID * NHID + NPAD * NHID) + 255) / 256, 256, 0,
             stream>>>(W0, Ws, Wout, w0t, wst, woutt);

  const int gemmBlocks = (MTILES + 3) / 4;  // 4 waves/block, 1 row-tile per wave
  k_gemm_in<<<gemmBlocks, 128, 0, stream>>>(xh, w0t, b0, x0, xcur);

  const int scatBlocks = (NNZP * NHID) / 256;  // 400000 exact
  for (int l = 0; l < NLAYER; ++l) {
    float beta = logf(0.5f / (float)(l + 1) + 1.0f);
    hipMemsetAsync(xe, 0, (size_t)NEDGE * NHID * sizeof(float), stream);
    k_scatter<<<scatBlocks, 256, 0, stream>>>(xcur, vertex, edges, xe);   // vertex -> edge
    k_edge_scale<<<(NEDGE * NHID + 255) / 256, 256, 0, stream>>>(xe, esc);
    hipMemsetAsync(xv, 0, (size_t)NV * NHID * sizeof(float), stream);
    k_scatter<<<scatBlocks, 256, 0, stream>>>(xe, edges, vertex, xv);     // edge -> vertex
    k_vertex_update<<<(NV + 7) / 8, 256, 0, stream>>>(xv, degV, x0, xih);
    k_gemm_layer<<<gemmBlocks, 128, 0, stream>>>(xih, wst + (size_t)l * NHID * NHID, xv,
                                                 beta, xcur, xfh);
  }

  k_gemm_out<<<gemmBlocks, 128, 0, stream>>>(xfh, woutt, bout, out);
  k_logsoftmax<<<(NV + 7) / 8, 256, 0, stream>>>(out);
}
